// WiKG_74560632259324
// MI455X (gfx1250) — matrix operand order repaired and checked by reading, not compiled, not run on hardware
//
#include <hip/hip_runtime.h>
#include <hip/hip_bf16.h>

// ---------------------------------------------------------------------------
// WiKG forward for MI455X (gfx1250, wave32, WMMA).
// Heavy GEMMs (logits 10k x 10k x 512, kv 60k x 1024 x 512, four 10k x 512 x 512)
// run on v_wmma_f32_16x16x32_bf16 (fp32 accumulate), with LDS double-buffered
// via global_load_async_to_lds_b128 (ASYNCcnt). Everything else is
// bandwidth-trivial vs 23.3 TB/s HBM.
// ---------------------------------------------------------------------------

#define N_NODES 10000
#define DIMV    512
#define HEADS   8
#define HEAD_DIM 64
#define TOPK    6

typedef __bf16 bf16x16 __attribute__((ext_vector_type(16)));
typedef float  floatx8 __attribute__((ext_vector_type(8)));

__device__ __forceinline__ unsigned short f2bf(float f) {
    union { float f; unsigned int u; } c; c.f = f;
    unsigned int u = c.u;
    return (unsigned short)((u + 0x7FFFu + ((u >> 16) & 1u)) >> 16);
}
__device__ __forceinline__ float bf2f(unsigned short h) {
    union { unsigned int u; float f; } c; c.u = ((unsigned int)h) << 16;
    return c.f;
}

// 32 bytes/lane, global -> LDS, ASYNCcnt-tracked (CDNA5 async copy path).
// INST_OFFSET applies to both LDS and global address (ISA 08 §4.4).
__device__ __forceinline__ void async_tile32B(unsigned lds_off, const void* g) {
    asm volatile(
        "global_load_async_to_lds_b128 %0, %1, off\n\t"
        "global_load_async_to_lds_b128 %0, %1, off offset:16"
        :: "v"(lds_off), "v"(g) : "memory");
}
__device__ __forceinline__ void wait_async0() {
    asm volatile("s_wait_asynccnt 0x0" ::: "memory");
}

// ------------------------------ tiny kernels --------------------------------

__global__ __launch_bounds__(256) void cast_bf16_k(const float* __restrict__ in,
                                                   unsigned short* __restrict__ out,
                                                   int n) {
    int i = blockIdx.x * 256 + threadIdx.x;
    if (i < n) out[i] = f2bf(in[i]);
}

__global__ __launch_bounds__(256) void zero_i32_k(int* p, int n) {
    int i = blockIdx.x * 256 + threadIdx.x;
    if (i < n) p[i] = 0;
}

__global__ __launch_bounds__(256) void colmean_k(const float* __restrict__ h,
                                                 float* __restrict__ mean, int M, int D) {
    __shared__ float red[256];
    int d = blockIdx.x, t = threadIdx.x;
    float s = 0.f;
    for (int r = t; r < M; r += 256) s += h[(size_t)r * D + d];
    red[t] = s; __syncthreads();
    for (int k = 128; k > 0; k >>= 1) { if (t < k) red[t] += red[t + k]; __syncthreads(); }
    if (t == 0) mean[d] = red[0] / (float)M;
}

__global__ __launch_bounds__(256) void havg_cast_k(float* __restrict__ h,
                                                   const float* __restrict__ mean,
                                                   unsigned short* __restrict__ hbf,
                                                   int total, int D) {
    int i = blockIdx.x * 256 + threadIdx.x;
    if (i < total) {
        float v = 0.5f * (h[i] + mean[i % D]);
        h[i] = v;
        hbf[i] = f2bf(v);
    }
}

__global__ __launch_bounds__(256) void bincount_k(const int* __restrict__ idx,
                                                  int* __restrict__ counts, int n) {
    int i = blockIdx.x * 256 + threadIdx.x;
    if (i < n) atomicAdd(&counts[idx[i]], 1);
}

__global__ __launch_bounds__(256) void argmax_k(const int* __restrict__ counts, int M,
                                                int* __restrict__ out) {
    __shared__ int sv[256], si[256];
    int t = threadIdx.x;
    int best = -1, bidx = 0;
    for (int i = t; i < M; i += 256) {
        int c = counts[i];
        if (c > best || (c == best && i < bidx)) { best = c; bidx = i; }
    }
    sv[t] = best; si[t] = bidx;
    __syncthreads();
    for (int s = 128; s > 0; s >>= 1) {
        if (t < s) {
            if (sv[t + s] > sv[t] || (sv[t + s] == sv[t] && si[t + s] < si[t])) {
                sv[t] = sv[t + s]; si[t] = si[t + s];
            }
        }
        __syncthreads();
    }
    if (t == 0) out[0] = si[0];
}

// ins_top = x[top_node]; q_heads = (h[top_node] @ q_w^T + q_b) * scale
__global__ __launch_bounds__(512) void topnode_q_k(const float* __restrict__ x,
                                                   const float* __restrict__ h,
                                                   const float* __restrict__ qw,
                                                   const float* __restrict__ qb,
                                                   const int* __restrict__ topn,
                                                   float* __restrict__ ins_top,
                                                   float* __restrict__ q_heads, int D) {
    __shared__ float qf[DIMV];
    int t = threadIdx.x;
    int tn = topn[0];
    ins_top[t] = x[(size_t)tn * D + t];
    qf[t] = h[(size_t)tn * D + t];
    __syncthreads();
    float s = qb[t];
    const float* wrow = qw + (size_t)t * D;
    for (int k = 0; k < D; ++k) s += qf[k] * wrow[k];
    q_heads[t] = s * 0.125f;  // HEAD_DIM^-0.5
}

// --------------------------- generic WMMA GEMM ------------------------------
// C[M x N] = act(A[M x K] @ B[N x K]^T + bias) [+ residual]
// A rows optionally remapped through row_map (fuses the nb = e_t[topk] gather).
// LDS double-buffered with async global->LDS copies: one barrier per k-step,
// next tile streams in while the current tile is in the matrix pipe.
__global__ __launch_bounds__(128)
void wmma_gemm_bf16(const unsigned short* __restrict__ A,
                    const unsigned short* __restrict__ B,
                    const float* __restrict__ bias,
                    float* __restrict__ Cf,
                    unsigned short* __restrict__ Cbf,
                    const int* __restrict__ row_map,
                    const float* __restrict__ residual,
                    int M, int N, int K, int leaky) {
    __shared__ __align__(16) unsigned short lsA[2][64 * 32];
    __shared__ __align__(16) unsigned short lsB[2][64 * 32];
    const int tid = threadIdx.x;
    const int wave = tid >> 5, lane = tid & 31;
    const int mBase = blockIdx.y * 64;
    const int nBase = blockIdx.x * 64;
    const int srow = tid >> 1;
    const int sk   = (tid & 1) * 16;
    const int nk   = K >> 5;

    floatx8 acc[4] = {};

    auto issue = [&](int kb, int b) {
        {   // A tile (64 rows x 32 k), optional row gather, OOB rows -> zeros
            int gr = mBase + srow;
            if (gr < M) {
                int ar = row_map ? row_map[gr] : gr;
                async_tile32B((unsigned)(size_t)(lsA[b] + srow * 32 + sk),
                              A + (size_t)ar * K + kb + sk);
            } else {
                uint4 z = {};
                uint4* q = (uint4*)(lsA[b] + srow * 32 + sk);
                q[0] = z; q[1] = z;
            }
        }
        {   // B tile (64 cols x 32 k)
            int gc = nBase + srow;
            if (gc < N) {
                async_tile32B((unsigned)(size_t)(lsB[b] + srow * 32 + sk),
                              B + (size_t)gc * K + kb + sk);
            } else {
                uint4 z = {};
                uint4* q = (uint4*)(lsB[b] + srow * 32 + sk);
                q[0] = z; q[1] = z;
            }
        }
    };

    issue(0, 0);
    for (int i = 0; i < nk; ++i) {
        const int buf = i & 1;
        wait_async0();        // this wave's async copies for tile i landed
        __syncthreads();      // every wave's copies landed; prev reads retired
        if (i + 1 < nk) issue((i + 1) << 5, buf ^ 1);

        // A fragment: lanes 0-15 -> rows, K 0-7 & 16-23; lanes 16-31 -> K 8-15 & 24-31
        union F { bf16x16 v; uint4 q[2]; } a, bf[4];
        {
            int row = wave * 16 + (lane & 15);
            int kh  = (lane & 16) ? 8 : 0;
            a.q[0] = *(const uint4*)(lsA[buf] + row * 32 + kh);
            a.q[1] = *(const uint4*)(lsA[buf] + row * 32 + kh + 16);
        }
        // B fragments hoisted so one DS wait covers all four WMMAs
        const int colL = lane & 15;
        const int khB  = (lane & 16) ? 8 : 0;
        #pragma unroll
        for (int nt = 0; nt < 4; ++nt) {
            bf[nt].q[0] = *(const uint4*)(lsB[buf] + (nt * 16 + colL) * 32 + khB);
            bf[nt].q[1] = *(const uint4*)(lsB[buf] + (nt * 16 + colL) * 32 + khB + 16);
        }
        #pragma unroll
        for (int nt = 0; nt < 4; ++nt)
            acc[nt] = __builtin_amdgcn_wmma_f32_16x16x32_bf16(
                false, a.v, false, bf[nt].v, (short)0, acc[nt], false, false);
    }

    // Epilogue. C layout: VGPR r, lane L -> (M = r + 8*(L>=16), N = L&15)
    const int rowOfs = (lane & 16) ? 8 : 0;
    const int colL = lane & 15;
    #pragma unroll
    for (int nt = 0; nt < 4; ++nt) {
        int col = nBase + nt * 16 + colL;
        float bv = bias ? bias[col] : 0.0f;
        #pragma unroll
        for (int r = 0; r < 8; ++r) {
            int row = mBase + wave * 16 + r + rowOfs;
            if (row < M) {
                float v = acc[nt][r] + bv;
                if (leaky) v = v > 0.0f ? v : v * 0.01f;
                size_t o = (size_t)row * N + col;
                if (residual) v += residual[o];
                if (Cf)  Cf[o]  = v;
                if (Cbf) Cbf[o] = f2bf(v);
            }
        }
    }
}

// ---------------------- fused logits + per-row top-6 ------------------------
// Block = 16 rows of e_h (resident in LDS), sweeps all 10000 columns in
// 64-wide WMMA chunks; 8 scanner threads per row keep sorted top-6 in regs.
// e_t tiles stream through a double-buffered async LDS pipeline.
__device__ __forceinline__ void topk_insert(float* tv, int* ti, float v, int j) {
    if (!(v > tv[5] || (v == tv[5] && j < ti[5]))) return;
    tv[5] = v; ti[5] = j;
    #pragma unroll
    for (int s = 5; s > 0; --s) {
        if (tv[s] > tv[s - 1] || (tv[s] == tv[s - 1] && ti[s] < ti[s - 1])) {
            float fv = tv[s]; tv[s] = tv[s - 1]; tv[s - 1] = fv;
            int   fi = ti[s]; ti[s] = ti[s - 1]; ti[s - 1] = fi;
        }
    }
}

__global__ __launch_bounds__(128)
void logits_topk_k(const unsigned short* __restrict__ EH,
                   const unsigned short* __restrict__ ET,
                   int* __restrict__ topk, int M, int K) {
    __shared__ __align__(16) unsigned short lsA[16 * DIMV];  // 16 rows resident
    __shared__ __align__(16) unsigned short lsB[2][64 * 32];
    __shared__ float lsC[16 * 64];
    __shared__ float mv[16 * 8 * 6];
    __shared__ int   mi[16 * 8 * 6];
    const int tid = threadIdx.x, wave = tid >> 5, lane = tid & 31;
    const int mBase = blockIdx.x * 16;

    auto issueB = [&](int jb, int kb, int b) {
        int gc = jb + (tid >> 1);
        int sk = (tid & 1) * 16;
        if (gc < M) {
            async_tile32B((unsigned)(size_t)(lsB[b] + (tid >> 1) * 32 + sk),
                          ET + (size_t)gc * K + kb + sk);
        } else {
            uint4 z = {};
            uint4* q = (uint4*)(lsB[b] + (tid >> 1) * 32 + sk);
            q[0] = z; q[1] = z;
        }
    };

    // stage the 16 A rows once (M = 10000 is an exact multiple of 16)
    for (int i = tid; i < 16 * DIMV / 8; i += 128)
        ((uint4*)lsA)[i] = ((const uint4*)EH)[(size_t)(mBase + (i >> 6)) * (DIMV / 8) + (i & 63)];

    float tv[6]; int ti[6];
    #pragma unroll
    for (int e = 0; e < 6; ++e) { tv[e] = -3.4e38f; ti[e] = 0x7fffffff; }
    const int row16 = tid >> 3;
    const int sub   = tid & 7;

    issueB(0, 0, 0);
    int buf = 0;
    for (int jb = 0; jb < M; jb += 64) {
        floatx8 acc = {};
        for (int kb = 0; kb < K; kb += 32) {
            wait_async0();
            __syncthreads();
            int nkb = kb + 32, njb = jb;
            if (nkb == K) { nkb = 0; njb += 64; }
            if (njb < M) issueB(njb, nkb, buf ^ 1);

            union F { bf16x16 v; uint4 q[2]; } a, b;
            {
                int row = lane & 15;
                int kh  = (lane & 16) ? 8 : 0;
                a.q[0] = *(const uint4*)(lsA + row * DIMV + kb + kh);
                a.q[1] = *(const uint4*)(lsA + row * DIMV + kb + kh + 16);
            }
            {
                int col = wave * 16 + (lane & 15);
                int kh  = (lane & 16) ? 8 : 0;
                b.q[0] = *(const uint4*)(lsB[buf] + col * 32 + kh);
                b.q[1] = *(const uint4*)(lsB[buf] + col * 32 + kh + 16);
            }
            acc = __builtin_amdgcn_wmma_f32_16x16x32_bf16(
                false, a.v, false, b.v, (short)0, acc, false, false);
            buf ^= 1;
        }
        {
            int rofs = (lane & 16) ? 8 : 0, c = lane & 15;
            #pragma unroll
            for (int r = 0; r < 8; ++r)
                lsC[(r + rofs) * 64 + wave * 16 + c] = acc[r];
        }
        __syncthreads();
        #pragma unroll
        for (int m = 0; m < 8; ++m) {
            int cc = sub * 8 + m;
            int j = jb + cc;
            if (j < M) topk_insert(tv, ti, lsC[row16 * 64 + cc], j);
        }
        __syncthreads();  // scan done before lsC is overwritten next sweep
    }

    // merge the 8 sorted partial lists per row
    int base = (row16 * 8 + sub) * 6;
    #pragma unroll
    for (int e = 0; e < 6; ++e) { mv[base + e] = tv[e]; mi[base + e] = ti[e]; }
    __syncthreads();
    if (tid < 16) {
        int ptr[8] = {0, 0, 0, 0, 0, 0, 0, 0};
        for (int e = 0; e < 6; ++e) {
            float best = -3.4e38f; int bidx = 0x7fffffff; int bs = 0;
            #pragma unroll
            for (int s = 0; s < 8; ++s) {
                if (ptr[s] < 6) {
                    float v = mv[(tid * 8 + s) * 6 + ptr[s]];
                    int  jj = mi[(tid * 8 + s) * 6 + ptr[s]];
                    if (v > best || (v == best && jj < bidx)) { best = v; bidx = jj; bs = s; }
                }
            }
            topk[(size_t)(mBase + tid) * TOPK + e] = bidx;
            ptr[bs]++;
        }
    }
}

// ------------------------------- attention ----------------------------------
__global__ __launch_bounds__(64)
void attention_k(const unsigned short* __restrict__ kv,
                 const float* __restrict__ qh,
                 float* __restrict__ out, int n_nodes) {
    __shared__ float att[HEADS][TOPK];
    __shared__ float p[HEADS][TOPK];
    const int n = blockIdx.x, t = threadIdx.x;
    const unsigned short* kvn = kv + (size_t)n * TOPK * 1024;
    if (t < HEADS * TOPK) {
        int h = t / TOPK, e = t % TOPK;
        const unsigned short* kp = kvn + e * 1024 + h * HEAD_DIM;
        float s = 0.f;
        for (int d = 0; d < HEAD_DIM; ++d) s += qh[h * HEAD_DIM + d] * bf2f(kp[d]);
        att[h][e] = s;
    }
    __syncthreads();
    if (t < HEADS) {
        float mx = att[t][0];
        for (int e = 1; e < TOPK; ++e) mx = fmaxf(mx, att[t][e]);
        float sum = 0.f;
        for (int e = 0; e < TOPK; ++e) { float ex = __expf(att[t][e] - mx); p[t][e] = ex; sum += ex; }
        float inv = 1.0f / sum;
        for (int e = 0; e < TOPK; ++e) p[t][e] *= inv;
    }
    __syncthreads();
    int h = t >> 3, d0 = (t & 7) * 8;
    #pragma unroll
    for (int i = 0; i < 8; ++i) {
        int d = d0 + i;
        float s = 0.f;
        #pragma unroll
        for (int e = 0; e < TOPK; ++e)
            s += p[h][e] * bf2f(kvn[e * 1024 + DIMV + h * HEAD_DIM + d]);
        out[(size_t)n * DIMV + h * HEAD_DIM + d] = s;
    }
}

// ----------------- fused LN(out) -> add e_h -> LN -> bf16 -------------------
__global__ __launch_bounds__(256)
void ln_fuse_k(const float* __restrict__ ao, const float* __restrict__ eh,
               const float* __restrict__ nw, const float* __restrict__ nb_,
               unsigned short* __restrict__ s_bf) {
    __shared__ float red[256];
    const int n = blockIdx.x, t = threadIdx.x;
    const size_t base = (size_t)n * DIMV;
    float v0 = ao[base + t], v1 = ao[base + t + 256];

    red[t] = v0 + v1; __syncthreads();
    for (int s = 128; s > 0; s >>= 1) { if (t < s) red[t] += red[t + s]; __syncthreads(); }
    float mu = red[0] / (float)DIMV; __syncthreads();
    float d0 = v0 - mu, d1 = v1 - mu;
    red[t] = d0 * d0 + d1 * d1; __syncthreads();
    for (int s = 128; s > 0; s >>= 1) { if (t < s) red[t] += red[t + s]; __syncthreads(); }
    float inv = rsqrtf(red[0] / (float)DIMV + 1e-5f); __syncthreads();
    float e0 = d0 * inv * nw[t] + nb_[t];
    float e1 = d1 * inv * nw[t + 256] + nb_[t + 256];

    float t0 = eh[base + t] + e0;
    float t1 = eh[base + t + 256] + e1;
    red[t] = t0 + t1; __syncthreads();
    for (int s = 128; s > 0; s >>= 1) { if (t < s) red[t] += red[t + s]; __syncthreads(); }
    float mu2 = red[0] / (float)DIMV; __syncthreads();
    float f0 = t0 - mu2, f1 = t1 - mu2;
    red[t] = f0 * f0 + f1 * f1; __syncthreads();
    for (int s = 128; s > 0; s >>= 1) { if (t < s) red[t] += red[t + s]; __syncthreads(); }
    float inv2 = rsqrtf(red[0] / (float)DIMV + 1e-5f);
    s_bf[base + t]       = f2bf(f0 * inv2 * nw[t] + nb_[t]);
    s_bf[base + t + 256] = f2bf(f1 * inv2 * nw[t + 256] + nb_[t + 256]);
}

// --------------------------------- launch -----------------------------------

extern "C" void kernel_launch(void* const* d_in, const int* in_sizes, int n_in,
                              void* d_out, int out_size, void* d_ws, size_t ws_size,
                              hipStream_t stream) {
    const float* x     = (const float*)d_in[0];
    const float* fc1_w = (const float*)d_in[1];
    const float* fc1_b = (const float*)d_in[2];
    const float* wh_w  = (const float*)d_in[3];
    const float* wh_b  = (const float*)d_in[4];
    const float* wt_w  = (const float*)d_in[5];
    const float* wt_b  = (const float*)d_in[6];
    const float* q_w   = (const float*)d_in[7];
    const float* q_b   = (const float*)d_in[8];
    const float* kv_w  = (const float*)d_in[9];
    const float* kv_b  = (const float*)d_in[10];
    const float* lin_w = (const float*)d_in[11];
    const float* lin_b = (const float*)d_in[12];
    const float* nrm_w = (const float*)d_in[13];
    const float* nrm_b = (const float*)d_in[14];

    const int N = N_NODES, D = DIMV;
    char* ws = (char*)d_ws;
    size_t off = 0;
    auto alloc = [&](size_t b) { size_t o = off; off += (b + 255) & ~(size_t)255; return o; };

    unsigned short* x_bf   = (unsigned short*)(ws + alloc((size_t)N * D * 2));
    unsigned short* fc1_bf = (unsigned short*)(ws + alloc((size_t)D * D * 2));
    unsigned short* whw_bf = (unsigned short*)(ws + alloc((size_t)D * D * 2));
    unsigned short* wtw_bf = (unsigned short*)(ws + alloc((size_t)D * D * 2));
    unsigned short* linw_bf= (unsigned short*)(ws + alloc((size_t)D * D * 2));
    unsigned short* kvw_bf = (unsigned short*)(ws + alloc((size_t)2 * D * D * 2));
    float* h       = (float*)(ws + alloc((size_t)N * D * 4));
    unsigned short* h_bf   = (unsigned short*)(ws + alloc((size_t)N * D * 2));
    float* e_h     = (float*)(ws + alloc((size_t)N * D * 4));
    float* e_t     = (float*)(ws + alloc((size_t)N * D * 4));
    unsigned short* eh_bf  = (unsigned short*)(ws + alloc((size_t)N * D * 2));
    unsigned short* et_bf  = (unsigned short*)(ws + alloc((size_t)N * D * 2));
    float* colmean = (float*)(ws + alloc((size_t)D * 4));
    int*   topk    = (int*)(ws + alloc((size_t)N * TOPK * 4));
    int*   counts  = (int*)(ws + alloc((size_t)N * 4));
    int*   topn    = (int*)(ws + alloc(256));
    float* q_heads = (float*)(ws + alloc((size_t)D * 4));
    unsigned short* kv_bf = (unsigned short*)(ws + alloc((size_t)N * TOPK * 2 * D * 2));
    float* attn_out= (float*)(ws + alloc((size_t)N * D * 4));
    unsigned short* s_bf = (unsigned short*)(ws + alloc((size_t)N * D * 2));

    float* emb_out = (float*)d_out;                 // N*D embedding
    float* ins_top = (float*)d_out + (size_t)N * D; // 512-element tail

    // bf16 copies
    cast_bf16_k<<<(N * D + 255) / 256, 256, 0, stream>>>(x, x_bf, N * D);
    cast_bf16_k<<<(D * D + 255) / 256, 256, 0, stream>>>(fc1_w, fc1_bf, D * D);
    cast_bf16_k<<<(D * D + 255) / 256, 256, 0, stream>>>(wh_w, whw_bf, D * D);
    cast_bf16_k<<<(D * D + 255) / 256, 256, 0, stream>>>(wt_w, wtw_bf, D * D);
    cast_bf16_k<<<(D * D + 255) / 256, 256, 0, stream>>>(lin_w, linw_bf, D * D);
    cast_bf16_k<<<(2 * D * D + 255) / 256, 256, 0, stream>>>(kv_w, kvw_bf, 2 * D * D);

    dim3 gNxD(D / 64, (N + 63) / 64);
    // h = leaky_relu(x @ fc1_w^T + b)
    wmma_gemm_bf16<<<gNxD, 128, 0, stream>>>(x_bf, fc1_bf, fc1_b, h, nullptr,
                                             nullptr, nullptr, N, D, D, 1);
    // h = (h + mean_n(h)) * 0.5, plus bf16 copy
    colmean_k<<<D, 256, 0, stream>>>(h, colmean, N, D);
    havg_cast_k<<<(N * D + 255) / 256, 256, 0, stream>>>(h, colmean, h_bf, N * D, D);
    // e_h, e_t (fp32 + bf16)
    wmma_gemm_bf16<<<gNxD, 128, 0, stream>>>(h_bf, whw_bf, wh_b, e_h, eh_bf,
                                             nullptr, nullptr, N, D, D, 0);
    wmma_gemm_bf16<<<gNxD, 128, 0, stream>>>(h_bf, wtw_bf, wt_b, e_t, et_bf,
                                             nullptr, nullptr, N, D, D, 0);
    // top-6 of (e_h @ e_t^T) per row — scale > 0 doesn't change ordering
    logits_topk_k<<<N / 16, 128, 0, stream>>>(eh_bf, et_bf, topk, N, D);
    // bincount + argmax -> top_node
    zero_i32_k<<<(N + 255) / 256, 256, 0, stream>>>(counts, N);
    bincount_k<<<(N * TOPK + 255) / 256, 256, 0, stream>>>(topk, counts, N * TOPK);
    argmax_k<<<1, 256, 0, stream>>>(counts, N, topn);
    // ins_top and q_heads
    topnode_q_k<<<1, 512, 0, stream>>>(x, h, q_w, q_b, topn, ins_top, q_heads, D);
    // kv = e_t[topk] @ kv_w^T + kv_b  (row gather fused via row_map), bf16 output
    dim3 gKV(2 * D / 64, (N * TOPK + 63) / 64);
    wmma_gemm_bf16<<<gKV, 128, 0, stream>>>(et_bf, kvw_bf, kv_b, nullptr, kv_bf,
                                            topk, nullptr, N * TOPK, 2 * D, D, 0);
    // per-node 8-head attention over the 6 neighbors
    attention_k<<<N, 64, 0, stream>>>(kv_bf, q_heads, attn_out, N);
    // s = LN(e_h + LN(out)) -> bf16
    ln_fuse_k<<<N, 256, 0, stream>>>(attn_out, e_h, nrm_w, nrm_b, s_bf);
    // embedding = leaky_relu(s @ lin_w^T + lin_b) + x  (straight into d_out)
    wmma_gemm_bf16<<<gNxD, 128, 0, stream>>>(s_bf, linw_bf, lin_b, emb_out, nullptr,
                                             nullptr, x, N, D, D, 1);
    (void)in_sizes; (void)n_in; (void)out_size; (void)ws_size;
}
